// TransformerBlock_33260226740787
// MI455X (gfx1250) — matrix operand order repaired and checked by reading, not compiled, not run on hardware
//
#include <hip/hip_runtime.h>
#include <hip/hip_bf16.h>

// ---------------------------------------------------------------------------
// MI455X (gfx1250) transformer block: all GEMMs via v_wmma_f32_16x16x32_f16.
// GEMM tile staging via TENSOR_LOAD_TO_LDS (TDM, TENSORcnt); attention K/V
// staging via GLOBAL_LOAD_ASYNC_TO_LDS_B128 (ASYNCcnt).
// B=2, T=4096, C=768, H=12, D=64, MLP=4x.  wave32.
// ---------------------------------------------------------------------------

typedef __attribute__((ext_vector_type(16))) _Float16     v16h;
typedef __attribute__((ext_vector_type(8)))  float        v8f;
typedef __attribute__((ext_vector_type(4)))  int          v4i;
typedef __attribute__((ext_vector_type(8)))  int          v8i;
typedef __attribute__((ext_vector_type(4)))  unsigned int v4u;

struct __align__(16) h8 { _Float16 v[8]; };

typedef __attribute__((address_space(1))) v4i*      gptr_v4i;
typedef __attribute__((address_space(3))) v4i*      lptr_v4i;
typedef __attribute__((address_space(3))) _Float16* lptr_f16;

#if __has_builtin(__builtin_amdgcn_global_load_async_to_lds_b128) && \
    __has_builtin(__builtin_amdgcn_s_wait_asynccnt)
#define HAVE_ASYNC_LDS 1
#else
#define HAVE_ASYNC_LDS 0
#endif

#if __has_builtin(__builtin_amdgcn_tensor_load_to_lds) && \
    __has_builtin(__builtin_amdgcn_s_wait_tensorcnt)
#define HAVE_TDM 1
#else
#define HAVE_TDM 0
#endif

// copy 8 halves (16B) global -> LDS (async when available)
static __device__ __forceinline__ void stage_b128(_Float16* lds_dst,
                                                  const _Float16* gsrc) {
#if HAVE_ASYNC_LDS
    v4i* g = (v4i*)(void*)gsrc;   // drop const, reinterpret pointee (generic AS)
    v4i* l = (v4i*)lds_dst;
    __builtin_amdgcn_global_load_async_to_lds_b128(
        (gptr_v4i)g, (lptr_v4i)l, 0, 0);
#else
    *(h8*)lds_dst = *(const h8*)gsrc;
#endif
}

static __device__ __forceinline__ void stage_wait() {
#if HAVE_ASYNC_LDS
    __builtin_amdgcn_s_wait_asynccnt(0);
#endif
}

#if HAVE_TDM
// Issue one TDM 2D tile load: `rows` x `rowlen_elems` f16 elements from a
// row-major tensor with `row_stride_elems` elements per row, packed
// contiguously into LDS at lds_dst.  D# per CDNA5 ISA.
static __device__ __forceinline__ void tdm_load_2d_f16(
    _Float16* lds_dst, const _Float16* gsrc,
    unsigned rowlen_elems, unsigned rows,
    unsigned row_stride_elems, unsigned tensor_rows) {
    const unsigned long long ga  = (unsigned long long)gsrc;  // flat==global
    const unsigned lds_addr = (unsigned)(unsigned long long)(lptr_f16)lds_dst;

    // group 0: count=1 | lds_addr[63:32] | global_addr[120:64] | type=2
    v4u g0;
    g0.x = 1u;
    g0.y = lds_addr;
    g0.z = (unsigned)ga;
    g0.w = (unsigned)((ga >> 32) & 0x01FFFFFFull) | (2u << 30);

    // group 1: wg_mask=0, data_size=1 (2B), no barrier/iterate/pad,
    // tensor_dim0/1, tile_dim0/1 (tile_dim2=0 -> 2D), dim0 stride.
    const unsigned long long s0  = row_stride_elems;
    const unsigned           td0 = row_stride_elems;
    const unsigned           td1 = tensor_rows;
    v8i g1;
    g1[0] = (int)(1u << 16);                                   // data_size=2B
    g1[1] = (int)((td0 & 0xFFFFu) << 16);                      // td0[15:0]
    g1[2] = (int)((td0 >> 16) | ((td1 & 0xFFFFu) << 16));      // td0 hi|td1 lo
    g1[3] = (int)((td1 >> 16) | ((rowlen_elems & 0xFFFFu) << 16)); // tile_dim0
    g1[4] = (int)(rows & 0xFFFFu);                             // tile_dim1
    g1[5] = (int)(s0 & 0xFFFFFFFFull);                         // stride0 lo
    g1[6] = (int)((s0 >> 32) & 0xFFFFull);                     // stride0 hi
    g1[7] = 0;

    // 6-arg form on this toolchain: (g0, g1, g2, g3, g4, cpol)
    __builtin_amdgcn_tensor_load_to_lds(g0, g1, (v4i)0, (v4i)0, (v8i)0, 0);
}
#endif

static __device__ __forceinline__ v8f wmma16x16x32(v16h a, v16h b, v8f c) {
    return __builtin_amdgcn_wmma_f32_16x16x32_f16(
        /*neg_a=*/false, a, /*neg_b=*/false, b,
        /*c_mod=*/(short)0, c, /*reuse_a=*/false, /*reuse_b=*/false);
}

// ---------------------------------------------------------------------------
// fp32 -> f16 conversion (weights)
// ---------------------------------------------------------------------------
__global__ void f32_to_f16_kernel(const float* __restrict__ in,
                                  _Float16* __restrict__ out, int n) {
    int i = blockIdx.x * 256 + threadIdx.x;
    if (i < n) out[i] = (_Float16)in[i];
}

// ---------------------------------------------------------------------------
// LayerNorm: fp32 [rows,768] -> f16 [rows,768]. One block (256 thr) per row.
// ---------------------------------------------------------------------------
__global__ __launch_bounds__(256)
void ln_kernel(const float* __restrict__ x, const float* __restrict__ w,
               const float* __restrict__ b, _Float16* __restrict__ out) {
    const int row = blockIdx.x;
    const int tid = threadIdx.x;
    const float* xr = x + (size_t)row * 768;

    float v0 = xr[tid], v1 = xr[tid + 256], v2 = xr[tid + 512];
    __shared__ float rs[256], rss[256];
    rs[tid]  = v0 + v1 + v2;
    rss[tid] = v0 * v0 + v1 * v1 + v2 * v2;
    __syncthreads();
#pragma unroll
    for (int off = 128; off > 0; off >>= 1) {
        if (tid < off) { rs[tid] += rs[tid + off]; rss[tid] += rss[tid + off]; }
        __syncthreads();
    }
    const float mean = rs[0] * (1.0f / 768.0f);
    const float var  = rss[0] * (1.0f / 768.0f) - mean * mean;
    const float inv  = rsqrtf(var + 1e-5f);

    _Float16* orow = out + (size_t)row * 768;
    orow[tid]       = (_Float16)(w[tid]       * (v0 - mean) * inv + b[tid]);
    orow[tid + 256] = (_Float16)(w[tid + 256] * (v1 - mean) * inv + b[tid + 256]);
    orow[tid + 512] = (_Float16)(w[tid + 512] * (v2 - mean) * inv + b[tid + 512]);
}

// ---------------------------------------------------------------------------
// WMMA GEMM:  out[M,N] = A[M,K] (f16, row-major) @ Bw[K,N] (f16, row-major)
// EPI 0: store f16 (no bias)
// EPI 1: store f32 = acc + bias[n] + resid[m,n]
// EPI 2: store f16 = gelu(acc + bias[n])        (exact erf gelu)
// Block: 256 thr = 8 waves. Tile BM=128, BN=128, BK=32, LDS double-buffered.
// Tiles staged by TDM (wave 0 issues one descriptor per tile) or async LDS.
// Wave (wM,wN) in 4x2 grid -> 32x64 patch = 2x4 wmma tiles.
// ---------------------------------------------------------------------------
template <int EPI>
__global__ __launch_bounds__(256)
void gemm_wmma_kernel(const _Float16* __restrict__ A,
                      const _Float16* __restrict__ Bw,
                      const float* __restrict__ bias,
                      const float* __restrict__ resid,
                      void* __restrict__ outv,
                      int M, int N, int K) {
    constexpr int BM = 128, BN = 128, BK = 32;
    __shared__ alignas(16) _Float16 As[2][BM * BK];
    __shared__ alignas(16) _Float16 Bs[2][BK * BN];

    const int tid    = threadIdx.x;
    const int lane   = tid & 31;
    const int wave   = tid >> 5;
    const int half16 = lane >> 4;      // 0: lanes 0-15, 1: lanes 16-31
    const int l15    = lane & 15;
    const int waveM  = wave >> 1;      // 0..3
    const int waveN  = wave & 1;       // 0..1
    const int bm0    = blockIdx.y * BM;
    const int bn0    = blockIdx.x * BN;

    v8f acc[2][4];
#pragma unroll
    for (int i = 0; i < 2; ++i)
#pragma unroll
        for (int j = 0; j < 4; ++j) acc[i][j] = (v8f)(0.0f);

    // stage one BMxBK A tile and BKxBN B tile into buffer `buf`
    auto stage_tiles = [&](int buf, int k0) {
#if HAVE_TDM
        if (wave == 0) {
            tdm_load_2d_f16(&As[buf][0], &A[(size_t)bm0 * K + k0],
                            BK, BM, (unsigned)K, (unsigned)M);
            tdm_load_2d_f16(&Bs[buf][0], &Bw[(size_t)k0 * N + bn0],
                            BN, BK, (unsigned)N, (unsigned)K);
        }
#else
#pragma unroll
        for (int i = tid; i < (BM * BK) / 8; i += 256) {  // 512 chunks
            int r = i >> 2;             // BK/8 = 4 chunks per row
            int c = (i & 3) << 3;
            stage_b128(&As[buf][r * BK + c],
                       &A[(size_t)(bm0 + r) * K + k0 + c]);
        }
#pragma unroll
        for (int i = tid; i < (BK * BN) / 8; i += 256) {  // 512 chunks
            int r = i >> 4;             // BN/8 = 16 chunks per row
            int c = (i & 15) << 3;
            stage_b128(&Bs[buf][r * BN + c],
                       &Bw[(size_t)(k0 + r) * N + bn0 + c]);
        }
#endif
    };
    auto stage_sync = [&]() {
#if HAVE_TDM
        if (wave == 0) __builtin_amdgcn_s_wait_tensorcnt(0);
#else
        stage_wait();
#endif
        __syncthreads();
    };

    const int nk = K / BK;
    stage_tiles(0, 0);
    stage_sync();

    for (int kc = 0; kc < nk; ++kc) {
        const int buf = kc & 1;
        if (kc + 1 < nk) stage_tiles(buf ^ 1, (kc + 1) * BK);  // overlap

        // --- build fragments from LDS ---
        v16h af[2], bf[4];
#pragma unroll
        for (int mi = 0; mi < 2; ++mi) {
            const int m0 = waveM * 32 + mi * 16;
            const _Float16* ap = &As[buf][(m0 + l15) * BK];
#pragma unroll
            for (int p = 0; p < 8; ++p) {
                int kk = ((p & 4) << 2) + (half16 << 3) + ((p & 3) << 1);
                af[mi][2 * p]     = ap[kk];
                af[mi][2 * p + 1] = ap[kk + 1];
            }
        }
#pragma unroll
        for (int ni = 0; ni < 4; ++ni) {
            const int n0 = waveN * 64 + ni * 16;
#pragma unroll
            for (int p = 0; p < 8; ++p) {
                int kk = ((p & 4) << 2) + (half16 << 3) + ((p & 3) << 1);
                bf[ni][2 * p]     = Bs[buf][kk * BN + n0 + l15];
                bf[ni][2 * p + 1] = Bs[buf][(kk + 1) * BN + n0 + l15];
            }
        }
#pragma unroll
        for (int mi = 0; mi < 2; ++mi)
#pragma unroll
            for (int ni = 0; ni < 4; ++ni)
                acc[mi][ni] = wmma16x16x32(af[mi], bf[ni], acc[mi][ni]);

        stage_sync();      // next-tile DMA landed; buf free for overwrite
    }

    // --- epilogue ---
#pragma unroll
    for (int mi = 0; mi < 2; ++mi) {
#pragma unroll
        for (int ni = 0; ni < 4; ++ni) {
            const int gmb = bm0 + waveM * 32 + mi * 16 + (half16 << 3);
            const int gn  = bn0 + waveN * 64 + ni * 16 + l15;
            float bval = 0.0f;
            if (EPI != 0) bval = bias[gn];
#pragma unroll
            for (int r = 0; r < 8; ++r) {
                const size_t idx = (size_t)(gmb + r) * N + gn;
                float v = acc[mi][ni][r];
                if (EPI == 0) {
                    ((_Float16*)outv)[idx] = (_Float16)v;
                } else if (EPI == 1) {
                    ((float*)outv)[idx] = v + bval + resid[idx];
                } else {  // EPI == 2: exact gelu
                    v += bval;
                    v = 0.5f * v * (1.0f + erff(v * 0.70710678118f));
                    ((_Float16*)outv)[idx] = (_Float16)v;
                }
            }
        }
    }
}

// ---------------------------------------------------------------------------
// Flash-style causal attention, all matmuls on WMMA.
// qkv: f16 [B*T, 2304] (q|k|v, per-head contiguous 64). y: f16 [B*T, 768].
// Grid: (T/64, B*H). Block: 128 thr = 4 waves, each wave owns 16 query rows.
// KV iterated in 32-key tiles async-staged (ASYNCcnt) into LDS.
// ---------------------------------------------------------------------------
__global__ __launch_bounds__(128)
void attn_kernel(const _Float16* __restrict__ qkv,
                 _Float16* __restrict__ y, int T) {
    __shared__ alignas(16) _Float16 Kt[32 * 64];
    __shared__ alignas(16) _Float16 Vt[32 * 64];
    __shared__ alignas(16) _Float16 Ps[4][16 * 32];

    const int qtile = blockIdx.x;
    const int bh    = blockIdx.y;
    const int b     = bh / 12;
    const int h     = bh % 12;
    const int tid   = threadIdx.x;
    const int wave  = tid >> 5;
    const int lane  = tid & 31;
    const int half16 = lane >> 4;
    const int l15    = lane & 15;

    const size_t rowbase = (size_t)b * T;
    const int    qbase   = qtile * 64 + wave * 16;

    // --- load Q fragments (16 rows x 64 head-dim = two 16x32 A-fragments) ---
    v16h qf[2];
    const _Float16* qptr = qkv + (rowbase + qbase + l15) * 2304 + h * 64;
#pragma unroll
    for (int c = 0; c < 2; ++c) {
#pragma unroll
        for (int p = 0; p < 8; ++p) {
            int kd = 32 * c + ((p & 4) << 2) + (half16 << 3) + ((p & 3) << 1);
            qf[c][2 * p]     = qptr[kd];
            qf[c][2 * p + 1] = qptr[kd + 1];
        }
    }

    v8f O[4];
#pragma unroll
    for (int t = 0; t < 4; ++t) O[t] = (v8f)(0.0f);
    float mrow[8], srow[8];
#pragma unroll
    for (int r = 0; r < 8; ++r) { mrow[r] = -1e30f; srow[r] = 0.0f; }

    const int nkv = 2 * qtile + 2;  // 32-key tiles covering keys <= qtile*64+63
    for (int t = 0; t < nkv; ++t) {
        const int kv0 = t * 32;
        // --- async K/V tile staging: 32x64 halves each, 16B chunks ---
#pragma unroll
        for (int i = tid; i < (32 * 64) / 8; i += 128) {  // 256 chunks
            int kr = i >> 3;
            int dc = (i & 7) << 3;
            const size_t base = (rowbase + kv0 + kr) * 2304 + h * 64 + dc;
            stage_b128(&Kt[kr * 64 + dc], &qkv[base + 768]);
            stage_b128(&Vt[kr * 64 + dc], &qkv[base + 1536]);
        }
        stage_wait();
        __syncthreads();

        // --- S = Q @ K^T for 2 key groups of 16 ---
        float sv[2][8];
#pragma unroll
        for (int j = 0; j < 2; ++j) {
            v8f S = (v8f)(0.0f);
#pragma unroll
            for (int c = 0; c < 2; ++c) {
                v16h bk;
#pragma unroll
                for (int p = 0; p < 8; ++p) {
                    int kl = ((p & 4) << 2) + (half16 << 3) + ((p & 3) << 1) + 32 * c;
                    const _Float16* kp = &Kt[(16 * j + l15) * 64 + kl];
                    bk[2 * p]     = kp[0];
                    bk[2 * p + 1] = kp[1];
                }
                S = wmma16x16x32(qf[c], bk, S);
            }
            const int key = kv0 + 16 * j + l15;
#pragma unroll
            for (int r = 0; r < 8; ++r) {
                const int qrow = qbase + r + (half16 << 3);
                float s = S[r] * 0.125f;  // 1/sqrt(64)
                sv[j][r] = (key <= qrow) ? s : -1e30f;
            }
        }

        // --- online softmax (row stats across the 16-lane half-wave) ---
        float tmax[8];
#pragma unroll
        for (int r = 0; r < 8; ++r) tmax[r] = fmaxf(sv[0][r], sv[1][r]);
#pragma unroll
        for (int off = 1; off < 16; off <<= 1)
#pragma unroll
            for (int r = 0; r < 8; ++r)
                tmax[r] = fmaxf(tmax[r], __shfl_xor(tmax[r], off, 32));

        float factor[8], psum[8];
#pragma unroll
        for (int r = 0; r < 8; ++r) {
            float mnew = fmaxf(mrow[r], tmax[r]);
            factor[r]  = __expf(mrow[r] - mnew);
            float p0 = (sv[0][r] > -1e29f) ? __expf(sv[0][r] - mnew) : 0.0f;
            float p1 = (sv[1][r] > -1e29f) ? __expf(sv[1][r] - mnew) : 0.0f;
            sv[0][r] = p0; sv[1][r] = p1;
            psum[r]  = p0 + p1;
            mrow[r]  = mnew;
        }
#pragma unroll
        for (int off = 1; off < 16; off <<= 1)
#pragma unroll
            for (int r = 0; r < 8; ++r)
                psum[r] += __shfl_xor(psum[r], off, 32);
#pragma unroll
        for (int r = 0; r < 8; ++r) srow[r] = srow[r] * factor[r] + psum[r];

        // rescale O, stash P (C-layout -> LDS -> A-layout)
#pragma unroll
        for (int tt = 0; tt < 4; ++tt)
#pragma unroll
            for (int r = 0; r < 8; ++r) O[tt][r] *= factor[r];
#pragma unroll
        for (int j = 0; j < 2; ++j)
#pragma unroll
            for (int r = 0; r < 8; ++r)
                Ps[wave][(r + (half16 << 3)) * 32 + 16 * j + l15] =
                    (_Float16)sv[j][r];

        // --- O += P(16x32) @ V(32x64) ---
        v16h pf;
#pragma unroll
        for (int p = 0; p < 8; ++p) {
            int kk = ((p & 4) << 2) + (half16 << 3) + ((p & 3) << 1);
            pf[2 * p]     = Ps[wave][l15 * 32 + kk];
            pf[2 * p + 1] = Ps[wave][l15 * 32 + kk + 1];
        }
#pragma unroll
        for (int tt = 0; tt < 4; ++tt) {
            v16h bv;
#pragma unroll
            for (int p = 0; p < 8; ++p) {
                int kl = ((p & 4) << 2) + (half16 << 3) + ((p & 3) << 1);
                bv[2 * p]     = Vt[kl * 64 + 16 * tt + l15];
                bv[2 * p + 1] = Vt[(kl + 1) * 64 + 16 * tt + l15];
            }
            O[tt] = wmma16x16x32(pf, bv, O[tt]);
        }
        __syncthreads();
    }

    // --- normalize + store y (f16) ---
#pragma unroll
    for (int tt = 0; tt < 4; ++tt) {
#pragma unroll
        for (int r = 0; r < 8; ++r) {
            const int qrow = qbase + r + (half16 << 3);
            float v = O[tt][r] / srow[r];
            y[(rowbase + qrow) * 768 + h * 64 + 16 * tt + l15] = (_Float16)v;
        }
    }
}

// ---------------------------------------------------------------------------
// Host-side orchestration
// ---------------------------------------------------------------------------
extern "C" void kernel_launch(void* const* d_in, const int* in_sizes, int n_in,
                              void* d_out, int out_size, void* d_ws, size_t ws_size,
                              hipStream_t stream) {
    const int T = 4096, Bb = 2, C = 768;
    const int BT = Bb * T;                 // 8192 rows
    const int C3 = 3 * C;                  // 2304
    const int C4 = 4 * C;                  // 3072

    const float* x      = (const float*)d_in[0];
    const float* ln1_w  = (const float*)d_in[1];
    const float* ln1_b  = (const float*)d_in[2];
    const float* w_qkv  = (const float*)d_in[3];
    const float* w_proj = (const float*)d_in[4];
    const float* b_proj = (const float*)d_in[5];
    const float* ln2_w  = (const float*)d_in[6];
    const float* ln2_b  = (const float*)d_in[7];
    const float* w1     = (const float*)d_in[8];
    const float* b1     = (const float*)d_in[9];
    const float* w2     = (const float*)d_in[10];
    const float* b2     = (const float*)d_in[11];

    // workspace carve-up (256B aligned)
    char*  ws  = (char*)d_ws;
    size_t off = 0;
    auto carve = [&](size_t bytes) -> void* {
        void* p = ws + off;
        off = (off + bytes + 255) & ~(size_t)255;
        return p;
    };
    _Float16* h16     = (_Float16*)carve((size_t)BT * C  * 2);
    _Float16* qkv16   = (_Float16*)carve((size_t)BT * C3 * 2);
    _Float16* y16     = (_Float16*)carve((size_t)BT * C  * 2);
    float*    x1      = (float*)   carve((size_t)BT * C  * 4);
    _Float16* h2_16   = (_Float16*)carve((size_t)BT * C  * 2);
    _Float16* g16     = (_Float16*)carve((size_t)BT * C4 * 2);
    _Float16* wqkv16  = (_Float16*)carve((size_t)C  * C3 * 2);
    _Float16* wproj16 = (_Float16*)carve((size_t)C  * C  * 2);
    _Float16* w1_16   = (_Float16*)carve((size_t)C  * C4 * 2);
    _Float16* w2_16   = (_Float16*)carve((size_t)C4 * C  * 2);

    auto cvt = [&](const float* src, _Float16* dst, int n) {
        f32_to_f16_kernel<<<(n + 255) / 256, 256, 0, stream>>>(src, dst, n);
    };
    cvt(w_qkv,  wqkv16,  C * C3);
    cvt(w_proj, wproj16, C * C);
    cvt(w1,     w1_16,   C * C4);
    cvt(w2,     w2_16,   C4 * C);

    // --- attention sub-block ---
    ln_kernel<<<BT, 256, 0, stream>>>(x, ln1_w, ln1_b, h16);

    gemm_wmma_kernel<0><<<dim3(C3 / 128, BT / 128), 256, 0, stream>>>(
        h16, wqkv16, nullptr, nullptr, qkv16, BT, C3, C);

    attn_kernel<<<dim3(T / 64, Bb * 12), 128, 0, stream>>>(qkv16, y16, T);

    gemm_wmma_kernel<1><<<dim3(C / 128, BT / 128), 256, 0, stream>>>(
        y16, wproj16, b_proj, x, x1, BT, C, C);

    // --- MLP sub-block ---
    ln_kernel<<<BT, 256, 0, stream>>>(x1, ln2_w, ln2_b, h2_16);

    gemm_wmma_kernel<2><<<dim3(C4 / 128, BT / 128), 256, 0, stream>>>(
        h2_16, w1_16, b1, nullptr, g16, BT, C4, C);

    gemm_wmma_kernel<1><<<dim3(C / 128, BT / 128), 256, 0, stream>>>(
        g16, w2_16, b2, x1, d_out, BT, C, C4);
}
